// TransHyperScale_41437844471841
// MI455X (gfx1250) — compile-verified, matrix operand order checked by reading, not run on hardware
//
#include <hip/hip_runtime.h>
#include <stdint.h>

// ---------------------------------------------------------------------------
// TransHyperScale forward for MI455X (gfx1250).
// bf16 WMMA GEMMs with double-buffered TDM (tensor_load_to_lds) tile
// streaming -- falling back to global_load_async_to_lds_b128 if the TDM
// builtin is unavailable -- wave32 neighborhood attention, fused epilogues.
//
// Workspace layout:
//   regA  [ 96MB]  f32 activation ping
//   regB  [ 96MB]  f32 activation pong
//   regN  [ 96MB]  bf16 LN / attention-out scratch (GEMM A operand)
//   regH  [ 96MB]  f32 post-attention residual h
//   regQ  [384MB]  f32 qkv | bf16 mlp-hidden | bf16 im2col
//   regW  [ 16MB]  bf16 transposed weights [N][K]
// Total ~784MB.
// ---------------------------------------------------------------------------

typedef __attribute__((ext_vector_type(16))) __bf16   v16bf;
typedef __attribute__((ext_vector_type(8)))  float    v8f;
typedef __attribute__((ext_vector_type(4)))  unsigned v4u;
typedef __attribute__((ext_vector_type(8)))  int      v8i;
typedef __attribute__((ext_vector_type(4)))  int      v4i;

#define DIMC 384
#define HEADS 6

#if __has_builtin(__builtin_amdgcn_tensor_load_to_lds)
#define USE_TDM 1
#else
#define USE_TDM 0
#endif

__device__ __forceinline__ __bf16 f2bf(float f) {
  unsigned u = __builtin_bit_cast(unsigned, f);
  unsigned r = (u + 0x7FFFu + ((u >> 16) & 1u)) >> 16;   // round-to-nearest-even
  unsigned short s = (unsigned short)r;
  return __builtin_bit_cast(__bf16, s);
}

// ---------------------------------------------------------------------------
// GEMM: C[M,N] = act(A[M,K] @ Bt[N,K]^T + bias) (+ residual)
// A, Bt bf16 in global memory. Block tile 128x128, K-step 32, 8 waves (4x2),
// each wave computes 2x4 WMMA 16x16 tiles. K-tiles streamed into LDS by the
// Tensor Data Mover (or async copies), double buffered.
// MODE: 0 = f32 out; 1 = f32 out + residual; 2 = leaky-ReLU + bf16 out.
// ---------------------------------------------------------------------------
#define BM 128
#define BN 128
#define BK 32
#define LDS_S 40          // bf16 elems per LDS row (64B data + 16B pad)

#if USE_TDM
// One 2D TDM descriptor per tile: tile_dim0 = 32 elems (64B row),
// tile_dim1 = 128 rows, tensor_dim0_stride = K elems, data_size = 2B,
// pad_enable with interval 16 DWORDs + pad 4 DWORDs -> LDS row stride 80B.
__device__ __forceinline__ void tdm_load_tile(const void* gaddr, unsigned lds_addr,
                                              unsigned Kelems, unsigned Mrows)
{
  uint64_t ga = (uint64_t)(uintptr_t)gaddr;
  v4u g0;
  g0[0] = 1u;                                              // count=1, user D#
  g0[1] = lds_addr;                                        // lds_addr
  g0[2] = (unsigned)(ga & 0xFFFFFFFFu);                    // global_addr[31:0]
  g0[3] = (unsigned)((ga >> 32) & 0x01FFFFFFu) | (2u << 30); // addr[56:32]|type=2
  v8i g1;
  g1[0] = (int)((1u << 16) |      // data_size = 2B
                (1u << 20) |      // pad_enable
                (3u << 22) |      // pad_interval: 16 DWORDs
                (3u << 25));      // pad_amount:   4 DWORDs (16B)
  g1[1] = (int)((Kelems & 0xFFFFu) << 16);                  // tensor_dim0 lo16
  g1[2] = (int)((Kelems >> 16) | ((Mrows & 0xFFFFu) << 16));// dim0 hi | dim1 lo
  g1[3] = (int)((Mrows >> 16) | (32u << 16));               // dim1 hi | tile_dim0
  g1[4] = (int)BM;                                          // tile_dim1=128, tile_dim2=0
  g1[5] = (int)Kelems;                                      // tensor_dim0_stride lo32
  g1[6] = 0;
  g1[7] = 0;
  v4i gz = {0, 0, 0, 0};
#if defined(__clang_major__) && (__clang_major__ >= 23)
  v8i gz8 = {0, 0, 0, 0, 0, 0, 0, 0};
  __builtin_amdgcn_tensor_load_to_lds(g0, g1, gz, gz, gz8, 0);
#else
  __builtin_amdgcn_tensor_load_to_lds(g0, g1, gz, gz, 0);
#endif
}
#endif

__device__ __forceinline__ void wait_tiles() {
#if USE_TDM
#if __has_builtin(__builtin_amdgcn_s_wait_tensorcnt)
  __builtin_amdgcn_s_wait_tensorcnt(0);
#else
  asm volatile("s_wait_tensorcnt 0x0" ::: "memory");
#endif
#else
  asm volatile("s_wait_asynccnt 0x0" ::: "memory");
#endif
}

template<int MODE>
__global__ __launch_bounds__(256)
void gemm_kernel(const __bf16* __restrict__ A, const __bf16* __restrict__ Bt,
                 const float* __restrict__ bias, const float* __restrict__ res,
                 void* __restrict__ Cv, int M, int N, int K)
{
  constexpr int AE = BM * LDS_S;       // 5120 bf16
  constexpr int BE = BN * LDS_S;       // 5120 bf16
  __shared__ __bf16 lds[2][AE + BE];   // 40KB total

  const int tid   = threadIdx.x;
  const int lane  = tid & 31;
  const int wid   = tid >> 5;
  const int wid_s = __builtin_amdgcn_readfirstlane(wid);  // scalar wave id
  const int wm    = wid & 3;                // wave row (32 rows each)
  const int wn    = wid >> 2;               // wave col (64 cols each)
  const int m0    = blockIdx.x * BM;
  const int n0    = blockIdx.y * BN;

  v8f acc[2][4] = {};

  const unsigned ldsBase[2] = { (unsigned)(uintptr_t)&lds[0][0],
                                (unsigned)(uintptr_t)&lds[1][0] };

  const int fr = lane & 15;                 // fragment row/col within 16
  const int kb = (lane >> 4) * 8;           // fragment K base (0 or 8)
  const int nk = K / BK;

  // Issue the two 128x64B tile loads for K-tile kt into buffer `base`.
  auto issue_tiles = [&](int kt, unsigned base) {
#if USE_TDM
    if (wid_s == 0)
      tdm_load_tile(A + (size_t)m0 * K + (size_t)kt * BK, base,
                    (unsigned)K, (unsigned)M);
    else if (wid_s == 4)
      tdm_load_tile(Bt + (size_t)n0 * K + (size_t)kt * BK, base + AE * 2,
                    (unsigned)K, (unsigned)N);
#else
    // Waves 0-3: A tile rows; waves 4-7: B tile rows. One 64B row per lane,
    // four b128 chunks folded into the shared LDS/global instruction offset.
    const unsigned ldk2  = (unsigned)K * 2u;
    const unsigned kbyte = (unsigned)(kt * BK * 2);
    const __bf16* src   = (wid_s < 4) ? A : Bt;
    const unsigned row0 = (wid_s < 4) ? (unsigned)m0 : (unsigned)n0;
    const unsigned lb   = base + ((wid_s < 4) ? 0u : (unsigned)(AE * 2));
    const unsigned rr   = (unsigned)((wid_s & 3) * 32 + lane);
    const unsigned go   = (row0 + rr) * ldk2 + kbyte;
    const unsigned lo   = lb + rr * (LDS_S * 2);
    asm volatile("global_load_async_to_lds_b128 %0, %1, %2"
                 :: "v"(lo), "v"(go), "s"(src) : "memory");
    asm volatile("global_load_async_to_lds_b128 %0, %1, %2 offset:16"
                 :: "v"(lo), "v"(go), "s"(src) : "memory");
    asm volatile("global_load_async_to_lds_b128 %0, %1, %2 offset:32"
                 :: "v"(lo), "v"(go), "s"(src) : "memory");
    asm volatile("global_load_async_to_lds_b128 %0, %1, %2 offset:48"
                 :: "v"(lo), "v"(go), "s"(src) : "memory");
#endif
  };

  issue_tiles(0, ldsBase[0]);

  for (int kt = 0; kt < nk; ++kt) {
    wait_tiles();                                      // tile kt landed (per wave)
    __syncthreads();                                   // publish to all waves
    if (kt + 1 < nk) issue_tiles(kt + 1, ldsBase[(kt + 1) & 1]);  // overlap

    const __bf16* sA = &lds[kt & 1][0];
    const __bf16* sB = &lds[kt & 1][AE];

    // Fragments per ISA 16-bit layout: lane<16 -> row fr, K kb..kb+7 (elems
    // 0-7) and 16+kb.. (elems 8-15); lane>=16 -> kb=8 halves.
    v16bf af[2], bfr[4];
#pragma unroll
    for (int mi = 0; mi < 2; mi++) {
      const __bf16* p = sA + (wm * 32 + mi * 16 + fr) * LDS_S + kb;
#pragma unroll
      for (int j = 0; j < 8; j++) { af[mi][j] = p[j]; af[mi][j + 8] = p[j + 16]; }
    }
#pragma unroll
    for (int ni = 0; ni < 4; ni++) {
      const __bf16* p = sB + (wn * 64 + ni * 16 + fr) * LDS_S + kb;
#pragma unroll
      for (int j = 0; j < 8; j++) { bfr[ni][j] = p[j]; bfr[ni][j + 8] = p[j + 16]; }
    }

#pragma unroll
    for (int mi = 0; mi < 2; mi++)
#pragma unroll
      for (int ni = 0; ni < 4; ni++)
        acc[mi][ni] = __builtin_amdgcn_wmma_f32_16x16x32_bf16(
            false, af[mi], false, bfr[ni], (short)0, acc[mi][ni], false, false);
  }

  // Epilogue: C layout VGPR r -> M = r + 8*(lane>=16), N = lane&15.
  const int cro = (lane >> 4) * 8;
  const int cco = lane & 15;
#pragma unroll
  for (int mi = 0; mi < 2; mi++) {
#pragma unroll
    for (int ni = 0; ni < 4; ni++) {
      int gn = n0 + wn * 64 + ni * 16 + cco;
      float bv = bias ? bias[gn] : 0.0f;
#pragma unroll
      for (int r2 = 0; r2 < 8; r2++) {
        int gm = m0 + wm * 32 + mi * 16 + cro + r2;
        float v = acc[mi][ni][r2] + bv;
        if (MODE == 1) v += res[(size_t)gm * N + gn];
        if (MODE == 2) v = (v > 0.0f) ? v : 0.2f * v;
        if (MODE == 2) ((__bf16*)Cv)[(size_t)gm * N + gn] = f2bf(v);
        else           ((float*)Cv)[(size_t)gm * N + gn] = v;
      }
    }
  }
}

static inline void gemm(hipStream_t st, const __bf16* A, const __bf16* Bt,
                        const float* bias, const float* res, void* C,
                        int M, int N, int K, int mode)
{
  dim3 g(M / BM, N / BN), blk(256);
  if (mode == 2)      gemm_kernel<2><<<g, blk, 0, st>>>(A, Bt, bias, res, C, M, N, K);
  else if (mode == 1) gemm_kernel<1><<<g, blk, 0, st>>>(A, Bt, bias, res, C, M, N, K);
  else                gemm_kernel<0><<<g, blk, 0, st>>>(A, Bt, bias, res, C, M, N, K);
}

// ---------------------------------------------------------------------------
// Weight prep: W[K][N] f32  ->  Wt[N][K] bf16 (output-indexed, coalesced out)
// ---------------------------------------------------------------------------
__global__ void wprep_kernel(const float* __restrict__ w, __bf16* __restrict__ wt,
                             int K, int N, size_t total)
{
  size_t idx = (size_t)blockIdx.x * blockDim.x + threadIdx.x;
  if (idx >= total) return;
  int k = (int)(idx % K);
  int n = (int)(idx / K);
  wt[idx] = f2bf(w[(size_t)k * N + n]);
}

// ---------------------------------------------------------------------------
// LayerNorm over C=384, one 128-thread block per token; bf16 output.
// ---------------------------------------------------------------------------
__global__ __launch_bounds__(128)
void ln_kernel(const float* __restrict__ in, __bf16* __restrict__ out,
               const float* __restrict__ s, const float* __restrict__ b)
{
  __shared__ float red[4];
  const size_t t = blockIdx.x;
  const float* x = in + t * DIMC;
  const int tx = threadIdx.x;
  float v0 = x[tx], v1 = x[tx + 128], v2 = x[tx + 256];
  float sum = v0 + v1 + v2;
#pragma unroll
  for (int m = 16; m >= 1; m >>= 1) sum += __shfl_xor(sum, m, 32);
  const int lane = tx & 31, w = tx >> 5;
  if (lane == 0) red[w] = sum;
  __syncthreads();
  float mu = (red[0] + red[1] + red[2] + red[3]) * (1.0f / DIMC);
  float d0 = v0 - mu, d1 = v1 - mu, d2 = v2 - mu;
  float sq = d0 * d0 + d1 * d1 + d2 * d2;
#pragma unroll
  for (int m = 16; m >= 1; m >>= 1) sq += __shfl_xor(sq, m, 32);
  __syncthreads();
  if (lane == 0) red[w] = sq;
  __syncthreads();
  float var = (red[0] + red[1] + red[2] + red[3]) * (1.0f / DIMC);
  float rs = rsqrtf(var + 1e-5f);
  __bf16* o = out + t * DIMC;
  o[tx]       = f2bf(d0 * rs * s[tx]       + b[tx]);
  o[tx + 128] = f2bf(d1 * rs * s[tx + 128] + b[tx + 128]);
  o[tx + 256] = f2bf(d2 * rs * s[tx + 256] + b[tx + 256]);
}

// ---------------------------------------------------------------------------
// Neighborhood attention: one wave per (pixel, head); lane owns 2 of 64 dims.
// qkv f32 layout per token: [3][HEADS][64]; out bf16 [token][384].
// ---------------------------------------------------------------------------
__global__ __launch_bounds__(32)
void natt_kernel(const float* __restrict__ qkv, const float* __restrict__ rpb,
                 __bf16* __restrict__ out, int Himg, int Wimg, int K)
{
  const int p    = blockIdx.x;
  const int head = blockIdx.y;
  const int lane = threadIdx.x;
  const int HW   = Himg * Wimg;
  const int bimg = p / HW;
  const int rem  = p % HW;
  const int i = rem / Wimg, j = rem % Wimg;
  const int r = K / 2;
  int si = i - r; si = si < 0 ? 0 : si; if (si > Himg - K) si = Himg - K;
  int sj = j - r; sj = sj < 0 ? 0 : sj; if (sj > Wimg - K) sj = Wimg - K;

  const float* qp = qkv + (size_t)p * 1152 + head * 64;
  const float q0 = qp[lane * 2]     * 0.125f;   // 64^-0.5
  const float q1 = qp[lane * 2 + 1] * 0.125f;

  const int KK = K * K;
  const int R2 = 2 * K - 1;
  float l0 = -1e30f, l1 = -1e30f;

  for (int t = 0; t < KK; t++) {
    int a = t / K, bb = t % K;
    int ki = si + a, kj = sj + bb;
    size_t kp = (size_t)bimg * HW + (size_t)ki * Wimg + kj;
    const float* kv = qkv + kp * 1152 + 384 + head * 64;
    float part = q0 * kv[lane * 2] + q1 * kv[lane * 2 + 1];
#pragma unroll
    for (int m = 16; m >= 1; m >>= 1) part += __shfl_xor(part, m, 32);
    int rel_i = ki - i + (K - 1), rel_j = kj - j + (K - 1);
    float logit = part + rpb[(head * R2 + rel_i) * R2 + rel_j];
    if ((t & 31) == lane) { if (t < 32) l0 = logit; else l1 = logit; }
  }

  float mx = fmaxf(l0, l1);
#pragma unroll
  for (int m = 16; m >= 1; m >>= 1) mx = fmaxf(mx, __shfl_xor(mx, m, 32));
  float e0 = __expf(l0 - mx);        // invalid slots underflow to 0
  float e1 = __expf(l1 - mx);
  float se = e0 + e1;
#pragma unroll
  for (int m = 16; m >= 1; m >>= 1) se += __shfl_xor(se, m, 32);
  float inv = 1.0f / se;

  float o0 = 0.0f, o1 = 0.0f;
  for (int t = 0; t < KK; t++) {
    float ew = __shfl((t < 32) ? e0 : e1, t & 31, 32);
    int a = t / K, bb = t % K;
    int ki = si + a, kj = sj + bb;
    size_t kp = (size_t)bimg * HW + (size_t)ki * Wimg + kj;
    const float* vv = qkv + kp * 1152 + 768 + head * 64;
    o0 += ew * vv[lane * 2];
    o1 += ew * vv[lane * 2 + 1];
  }
  __bf16* op = out + (size_t)p * DIMC + head * 64;
  op[lane * 2]     = f2bf(o0 * inv);
  op[lane * 2 + 1] = f2bf(o1 * inv);
}

// ---------------------------------------------------------------------------
// im2col (3x3 stride-2 pad-1, NHWC f32 in) -> bf16 [B*OH*OW][9*384]
// ---------------------------------------------------------------------------
__global__ void im2col_kernel(const float* __restrict__ in, __bf16* __restrict__ out,
                              int Hin, int Win, size_t total)
{
  size_t idx = (size_t)blockIdx.x * blockDim.x + threadIdx.x;
  if (idx >= total) return;
  int kk = (int)(idx % 3456);
  size_t m = idx / 3456;
  int ci = kk % 384, kx = (kk / 384) % 3, ky = kk / 1152;
  int OW = Win / 2, OH = Hin / 2;
  int ox = (int)(m % OW);
  int oy = (int)((m / OW) % OH);
  int bb = (int)(m / ((size_t)OW * OH));
  int iy = 2 * oy - 1 + ky, ix = 2 * ox - 1 + kx;
  float v = 0.0f;
  if (iy >= 0 && iy < Hin && ix >= 0 && ix < Win)
    v = in[(((size_t)bb * Hin + iy) * Win + ix) * DIMC + ci];
  out[idx] = f2bf(v);
}

// ---------------------------------------------------------------------------
// Layout transposes (f32)
// ---------------------------------------------------------------------------
__global__ void nchw2nhwc_kernel(const float* __restrict__ in, float* __restrict__ out,
                                 int H, int W, size_t total)
{
  size_t idx = (size_t)blockIdx.x * blockDim.x + threadIdx.x;
  if (idx >= total) return;
  int c = (int)(idx % DIMC);
  size_t tkn = idx / DIMC;
  int xw = (int)(tkn % W);
  int y  = (int)((tkn / W) % H);
  int bb = (int)(tkn / ((size_t)W * H));
  out[idx] = in[(((size_t)bb * DIMC + c) * H + y) * W + xw];
}

__global__ void nhwc2nchw_kernel(const float* __restrict__ in, float* __restrict__ out,
                                 int H, int W, size_t total)
{
  size_t idx = (size_t)blockIdx.x * blockDim.x + threadIdx.x;
  if (idx >= total) return;
  int xw = (int)(idx % W);
  size_t t2 = idx / W;
  int y = (int)(t2 % H);
  size_t t3 = t2 / H;
  int c  = (int)(t3 % DIMC);
  int bb = (int)(t3 / DIMC);
  out[idx] = in[(((size_t)bb * H + y) * W + xw) * DIMC + c];
}

// ---------------------------------------------------------------------------
// Host orchestration
// ---------------------------------------------------------------------------
extern "C" void kernel_launch(void* const* d_in, const int* in_sizes, int n_in,
                              void* d_out, int out_size, void* d_ws, size_t ws_size,
                              hipStream_t stream)
{
  (void)in_sizes; (void)n_in; (void)out_size; (void)ws_size;
  const float* x = (const float*)d_in[0];

  char* ws = (char*)d_ws;
  const size_t MB = 1ull << 20;
  float*  regA = (float*)(ws);
  float*  regB = (float*)(ws + 96 * MB);
  __bf16* regN = (__bf16*)(ws + 192 * MB);
  float*  regH = (float*)(ws + 288 * MB);
  char*   regQ = (ws + 384 * MB);            // f32 qkv / bf16 hidden / bf16 im2col
  __bf16* regW = (__bf16*)(ws + 768 * MB);   // transposed bf16 weights

  // ---- one-time weight convert+transpose into regW ----
  __bf16* wt_qkv[3]; __bf16* wt_proj[3]; __bf16* wt_w1[3]; __bf16* wt_w2[3];
  __bf16* wt_conv[2];
  {
    size_t off = 0;
    auto prep = [&](const float* w, int K, int N) -> __bf16* {
      __bf16* dst = regW + off;
      size_t total = (size_t)K * N;
      off += total;
      wprep_kernel<<<(unsigned)((total + 255) / 256), 256, 0, stream>>>(
          w, dst, K, N, total);
      return dst;
    };
    for (int s = 0; s < 3; s++) {
      const int base = 1 + s * 12;
      wt_qkv[s]  = prep((const float*)d_in[base + 2],  384, 1152);
      wt_proj[s] = prep((const float*)d_in[base + 4],  384, 384);
      wt_w1[s]   = prep((const float*)d_in[base + 8],  384, 1536);
      wt_w2[s]   = prep((const float*)d_in[base + 10], 1536, 384);
    }
    wt_conv[0] = prep((const float*)d_in[37], 3456, 384);
    wt_conv[1] = prep((const float*)d_in[39], 3456, 384);
  }

  // ---- NCHW -> NHWC ----
  {
    size_t total = 4ull * DIMC * 128 * 128;
    nchw2nhwc_kernel<<<(unsigned)((total + 255) / 256), 256, 0, stream>>>(
        x, regA, 128, 128, total);
  }

  float* cur = regA;
  float* alt = regB;
  int Hs = 128;
  const int Ks[3] = {7, 5, 3};

  for (int s = 0; s < 3; s++) {
    const int base = 1 + s * 12;
    const float* ln1_s  = (const float*)d_in[base + 0];
    const float* ln1_b  = (const float*)d_in[base + 1];
    const float* rpb    = (const float*)d_in[base + 3];
    const float* proj_b = (const float*)d_in[base + 5];
    const float* ln2_s  = (const float*)d_in[base + 6];
    const float* ln2_b  = (const float*)d_in[base + 7];
    const float* b1     = (const float*)d_in[base + 9];
    const float* b2     = (const float*)d_in[base + 11];

    const int Ntok = 4 * Hs * Hs;

    // --- attention half ---
    ln_kernel<<<Ntok, 128, 0, stream>>>(cur, regN, ln1_s, ln1_b);
    gemm(stream, regN, wt_qkv[s], nullptr, nullptr, regQ, Ntok, 1152, 384, 0);
    {
      dim3 ag(Ntok, HEADS);
      natt_kernel<<<ag, 32, 0, stream>>>((const float*)regQ, rpb, regN, Hs, Hs, Ks[s]);
    }
    gemm(stream, regN, wt_proj[s], proj_b, cur, regH, Ntok, 384, 384, 1); // h = x+attn

    // --- MLP half ---
    ln_kernel<<<Ntok, 128, 0, stream>>>(regH, regN, ln2_s, ln2_b);
    gemm(stream, regN, wt_w1[s], b1, nullptr, regQ, Ntok, 1536, 384, 2);      // leaky->bf16
    gemm(stream, (const __bf16*)regQ, wt_w2[s], b2, regH, alt, Ntok, 384, 1536, 1);
    { float* t = cur; cur = alt; alt = t; }

    // --- downsample conv ---
    if (s < 2) {
      const float* db = (const float*)d_in[38 + s * 2];
      const int OH = Hs / 2;
      const int Mc = 4 * OH * OH;
      size_t total = (size_t)Mc * 3456;
      im2col_kernel<<<(unsigned)((total + 255) / 256), 256, 0, stream>>>(
          cur, (__bf16*)regQ, Hs, Hs, total);
      gemm(stream, (const __bf16*)regQ, wt_conv[s], db, nullptr, alt, Mc, 384, 3456, 0);
      { float* t = cur; cur = alt; alt = t; }
      Hs = OH;
    }
  }

  // ---- NHWC -> NCHW into d_out (Hs == 32) ----
  {
    size_t total = 4ull * DIMC * 32 * 32;
    nhwc2nchw_kernel<<<(unsigned)((total + 255) / 256), 256, 0, stream>>>(
        cur, (float*)d_out, 32, 32, total);
  }
}
